// HindiCausalLMAttention_75393855914580
// MI455X (gfx1250) — matrix operand order repaired and checked by reading, not compiled, not run on hardware
//
#include <hip/hip_runtime.h>
#include <hip/hip_bf16.h>

typedef __bf16 bf16;
typedef __attribute__((ext_vector_type(8)))  bf16  v8bf;
typedef __attribute__((ext_vector_type(16))) bf16  v16bf;
typedef __attribute__((ext_vector_type(8)))  float v8f;
typedef int v4i __attribute__((vector_size(16)));

#define CAT16(lo, hi) __builtin_shufflevector((lo), (hi), 0,1,2,3,4,5,6,7,8,9,10,11,12,13,14,15)

#define S_LEN 2048
#define D_MODEL 2048
#define HD 128
#define NH 16
#define NB 2

#if defined(__AMDGCN__) && __has_builtin(__builtin_amdgcn_global_load_async_to_lds_b128) && \
    __has_builtin(__builtin_amdgcn_s_wait_asynccnt)
#define USE_ASYNC_K 1
#define AS1P(p) ((__attribute__((address_space(1))) v4i*)(p))
#define AS3P(p) ((__attribute__((address_space(3))) v4i*)(p))
#else
#define USE_ASYNC_K 0
#endif

// ---------------------------------------------------------------------------
// Kernel 0: one-time f32 -> bf16 conversion (X and the four weight matrices).
// Removes all cvt work from GEMM inner loops and halves operand bandwidth.
// ---------------------------------------------------------------------------
__global__ __launch_bounds__(256)
void cvt_bf16_kernel(const float* __restrict__ src, bf16* __restrict__ dst) {
    const size_t i = ((size_t)blockIdx.x * 256 + threadIdx.x) * 8;
    const float4 f0 = *(const float4*)(src + i);
    const float4 f1 = *(const float4*)(src + i + 4);
    v8bf o;
    o[0] = (bf16)f0.x; o[1] = (bf16)f0.y; o[2] = (bf16)f0.z; o[3] = (bf16)f0.w;
    o[4] = (bf16)f1.x; o[5] = (bf16)f1.y; o[6] = (bf16)f1.z; o[7] = (bf16)f1.w;
    *(v8bf*)(dst + i) = o;
}

// ---------------------------------------------------------------------------
// Kernel 1: fused QKV projection, all-bf16 operands.  out[n,d] = X[n,:]·W[d,:]
// 32x64 tile per wave: 2 A-frags x 4 B-frags -> 8 WMMAs / K-step.
// Q,K stored head-major [bh][s][j]; V stored transposed [bh][j][s].
// ---------------------------------------------------------------------------
__global__ __launch_bounds__(256)
void qkv_kernel(const bf16* __restrict__ Xb,
                const bf16* __restrict__ Wq,
                const bf16* __restrict__ Wk,
                const bf16* __restrict__ Wv,
                bf16* __restrict__ Qh, bf16* __restrict__ Kh, bf16* __restrict__ Vt) {
    const int wave = threadIdx.x >> 5;
    const int lane = threadIdx.x & 31;
    const int lhi  = lane >> 4;
    const int llo  = lane & 15;

    const int tile = blockIdx.x * 8 + wave;   // 128 row-tiles x 96 col-tiles
    const int tr = tile % 128;                // 32-row tile
    const int tc = tile / 128;                // 64-col tile, 0..95
    const int sec = tc >> 5;                  // 0=Q 1=K 2=V
    const int dt  = (tc & 31) * 64;

    const bf16* W = (sec == 0) ? Wq : ((sec == 1) ? Wk : Wv);
    const bf16* arow0 = Xb + (size_t)(tr * 32 + llo) * D_MODEL;
    const bf16* arow1 = arow0 + 16 * D_MODEL;
    const bf16* wrow0 = W + (size_t)(dt + llo) * D_MODEL;

    v8f c[2][4] = {};
    for (int k0 = 0; k0 < D_MODEL; k0 += 32) {
        const int ka = k0 + lhi * 8;                   // A: {8 elems, gap, 8 elems}
        v16bf a0 = CAT16(*(const v8bf*)(arow0 + ka), *(const v8bf*)(arow0 + ka + 16));
        v16bf a1 = CAT16(*(const v8bf*)(arow1 + ka), *(const v8bf*)(arow1 + ka + 16));
        const int kb = k0 + lhi * 8;                   // B: {8 elems, gap, 8 elems}
        __builtin_prefetch(wrow0 + k0 + 512, 0, 1);
        for (int n = 0; n < 4; ++n) {
            v16bf b = CAT16(*(const v8bf*)(wrow0 + (size_t)n * 16 * D_MODEL + kb), *(const v8bf*)(wrow0 + (size_t)n * 16 * D_MODEL + kb + 16));
            c[0][n] = __builtin_amdgcn_wmma_f32_16x16x32_bf16(false, a0, false, b,
                                                              (short)0, c[0][n], false, false);
            c[1][n] = __builtin_amdgcn_wmma_f32_16x16x32_bf16(false, a1, false, b,
                                                              (short)0, c[1][n], false, false);
        }
    }
    for (int rr = 0; rr < 2; ++rr) {
        for (int n = 0; n < 4; ++n) {
            for (int r = 0; r < 8; ++r) {
                const int nrow = tr * 32 + rr * 16 + r + lhi * 8;
                const int bb = nrow >> 11;
                const int s  = nrow & (S_LEN - 1);
                const int col = dt + n * 16 + llo;
                const int h = col >> 7, j = col & (HD - 1);
                const bf16 val = (bf16)c[rr][n][r];
                if (sec == 0)
                    Qh[(((size_t)(bb * NH + h)) * S_LEN + s) * HD + j] = val;
                else if (sec == 1)
                    Kh[(((size_t)(bb * NH + h)) * S_LEN + s) * HD + j] = val;
                else
                    Vt[(((size_t)(bb * NH + h)) * HD + j) * S_LEN + s] = val;
            }
        }
    }
}

// ---------------------------------------------------------------------------
// Kernel 2: RoPE applied in place to Qh, Kh (position id == s after clamp).
// ---------------------------------------------------------------------------
__global__ __launch_bounds__(256)
void rope_kernel(bf16* __restrict__ Qh, bf16* __restrict__ Kh,
                 const float* __restrict__ cosT, const float* __restrict__ sinT) {
    const int idx = blockIdx.x * 256 + threadIdx.x;   // BH * S * 64
    const int j  = idx & 63;
    const int s  = (idx >> 6) & (S_LEN - 1);
    const int bh = idx >> 17;
    const size_t base = ((size_t)bh * S_LEN + s) * HD;
    const float c1 = cosT[s * HD + j],      s1 = sinT[s * HD + j];
    const float c2 = cosT[s * HD + j + 64], s2 = sinT[s * HD + j + 64];
    const float q1 = (float)Qh[base + j], q2 = (float)Qh[base + j + 64];
    Qh[base + j]      = (bf16)(q1 * c1 - q2 * s1);
    Qh[base + j + 64] = (bf16)(q2 * c2 + q1 * s2);
    const float k1 = (float)Kh[base + j], k2 = (float)Kh[base + j + 64];
    Kh[base + j]      = (bf16)(k1 * c1 - k2 * s1);
    Kh[base + j + 64] = (bf16)(k2 * c2 + k1 * s2);
}

// ---------------------------------------------------------------------------
// Kernel 3: causal flash attention. 4 waves/block, one 16-row q tile per wave,
// kv blocks of 32, online softmax, bf16 WMMA throughout.
// Next K tile is double-buffered into per-wave LDS via async global->LDS
// copies (ASYNCcnt path) when the toolchain exposes the builtin.
// ---------------------------------------------------------------------------
__global__ __launch_bounds__(128)
void fa_kernel(const bf16* __restrict__ Qh, const bf16* __restrict__ Kh,
               const bf16* __restrict__ Vt, bf16* __restrict__ Aout) {
    __shared__ alignas(16) bf16 plds[4][16 * 32];
#if USE_ASYNC_K
    __shared__ alignas(16) bf16 kstage[4][2][32 * HD];   // 8KB per buffer per wave
#endif
    const int wave = threadIdx.x >> 5;
    const int lane = threadIdx.x & 31;
    const int lhi = lane >> 4, llo = lane & 15;
    const int bh = blockIdx.x >> 5;          // 32 q-tiles of 64 per bh
    const int qt = blockIdx.x & 31;
    const int b = bh >> 4, h = bh & 15;
    const int qbase = qt * 64 + wave * 16;

    const bf16* Qb = Qh + (size_t)bh * S_LEN * HD;
    const bf16* Kb = Kh + (size_t)bh * S_LEN * HD;
    const bf16* Vb = Vt + (size_t)bh * HD * S_LEN;

    v16bf aq[4];
    {
        const bf16* qrow = Qb + (size_t)(qbase + llo) * HD;
        for (int f = 0; f < 4; ++f) {
            const int k0 = f * 32 + lhi * 8;
            aq[f] = CAT16(*(const v8bf*)(qrow + k0), *(const v8bf*)(qrow + k0 + 16));
        }
    }

    v8f o[8] = {};
    float rowmax[8], rowsum[8];
    for (int r = 0; r < 8; ++r) { rowmax[r] = -3.0e38f; rowsum[r] = 0.0f; }
    const float scale = 0.08838834764831845f;   // 1/sqrt(128)
    const int kbmax = (qbase + 15) >> 5;

#if USE_ASYNC_K
    {   // stage K tile for kb = 0 (16 x b128 per wave = 8KB)
        const bf16* src = Kb;
        bf16* dst = kstage[wave][0];
        for (int i = 0; i < 16; ++i) {
            const int off = (i * 32 + lane) * 8;
            __builtin_amdgcn_global_load_async_to_lds_b128(AS1P(src + off), AS3P(dst + off), 0, 0);
        }
    }
#endif

    for (int kb = 0; kb <= kbmax; ++kb) {
#if USE_ASYNC_K
        if (kb < kbmax) {   // prefetch next tile, then wait for current one
            const bf16* src = Kb + (size_t)(kb + 1) * 32 * HD;
            bf16* dst = kstage[wave][(kb + 1) & 1];
            for (int i = 0; i < 16; ++i) {
                const int off = (i * 32 + lane) * 8;
                __builtin_amdgcn_global_load_async_to_lds_b128(AS1P(src + off), AS3P(dst + off), 0, 0);
            }
            __builtin_amdgcn_s_wait_asynccnt(16);
        } else {
            __builtin_amdgcn_s_wait_asynccnt(0);
        }
        const bf16* ktile = kstage[wave][kb & 1];
#endif
        // ---- scores: 16 q-rows x 32 kv-cols --------------------------------
        v8f s[2];
        for (int hh = 0; hh < 2; ++hh) {
            v8f acc = {};
#if USE_ASYNC_K
            const bf16* krow = ktile + (size_t)(hh * 16 + llo) * HD;
#else
            const bf16* krow = Kb + (size_t)(kb * 32 + hh * 16 + llo) * HD;
#endif
            for (int f = 0; f < 4; ++f) {
                const int j0 = f * 32 + lhi * 8;
                v16bf bk = CAT16(*(const v8bf*)(krow + j0), *(const v8bf*)(krow + j0 + 16));
                acc = __builtin_amdgcn_wmma_f32_16x16x32_bf16(false, aq[f], false, bk,
                                                              (short)0, acc, false, false);
            }
            s[hh] = acc;
        }
        // ---- causal mask + online softmax ----------------------------------
        float corr[8], p0v[8], p1v[8];
        for (int r = 0; r < 8; ++r) {
            const int qrow = qbase + r + lhi * 8;
            const int c0 = kb * 32 + llo;
            float v0 = (c0      <= qrow) ? s[0][r] * scale : -3.0e38f;
            float v1 = (c0 + 16 <= qrow) ? s[1][r] * scale : -3.0e38f;
            float m = fmaxf(v0, v1);
            m = fmaxf(m, __shfl_xor(m, 1));
            m = fmaxf(m, __shfl_xor(m, 2));
            m = fmaxf(m, __shfl_xor(m, 4));
            m = fmaxf(m, __shfl_xor(m, 8));
            m = fmaxf(m, rowmax[r]);
            const float p0 = __expf(v0 - m);
            const float p1 = __expf(v1 - m);
            float su = p0 + p1;
            su += __shfl_xor(su, 1);
            su += __shfl_xor(su, 2);
            su += __shfl_xor(su, 4);
            su += __shfl_xor(su, 8);
            corr[r] = __expf(rowmax[r] - m);
            rowsum[r] = rowsum[r] * corr[r] + su;
            rowmax[r] = m;
            p0v[r] = p0; p1v[r] = p1;
        }
        // ---- P: C-layout -> A-layout via per-wave LDS ----------------------
        bf16* pl = plds[wave];
        for (int r = 0; r < 8; ++r) {
            const int mrow = r + lhi * 8;
            pl[mrow * 32 + llo]      = (bf16)p0v[r];
            pl[mrow * 32 + 16 + llo] = (bf16)p1v[r];
        }
        v16bf ap;
        {
            const bf16* prow = pl + llo * 32;
            ap = CAT16(*(const v8bf*)(prow + lhi * 8), *(const v8bf*)(prow + lhi * 8 + 16));
        }
        // ---- rescale accumulator, then PV ----------------------------------
        for (int c = 0; c < 8; ++c)
            for (int r = 0; r < 8; ++r) o[c][r] *= corr[r];
        for (int c = 0; c < 8; ++c) {
            const bf16* vrow = Vb + (size_t)(c * 16 + llo) * S_LEN + kb * 32 + lhi * 8;
            v16bf bv = CAT16(*(const v8bf*)(vrow), *(const v8bf*)(vrow + 16));
            o[c] = __builtin_amdgcn_wmma_f32_16x16x32_bf16(false, ap, false, bv,
                                                           (short)0, o[c], false, false);
        }
    }
    // ---- normalize + store to [n, d] bf16 for the output GEMM --------------
    for (int r = 0; r < 8; ++r) {
        const float inv = 1.0f / rowsum[r];
        const int nrow = b * S_LEN + qbase + r + lhi * 8;
        bf16* orow = Aout + (size_t)nrow * D_MODEL + h * HD;
        for (int c = 0; c < 8; ++c)
            orow[c * 16 + llo] = (bf16)(o[c][r] * inv);
    }
}

// ---------------------------------------------------------------------------
// Kernel 4: output projection  Out[n,d] = A[n,:]·Wo[d,:]  (bf16 x bf16 -> f32)
// 32x64 tile per wave, same structure as qkv_kernel.
// ---------------------------------------------------------------------------
__global__ __launch_bounds__(256)
void oproj_kernel(const bf16* __restrict__ A, const bf16* __restrict__ Wo,
                  float* __restrict__ Out) {
    const int wave = threadIdx.x >> 5;
    const int lane = threadIdx.x & 31;
    const int lhi = lane >> 4, llo = lane & 15;
    const int tile = blockIdx.x * 8 + wave;   // 128 row-tiles x 32 col-tiles
    const int tr = tile % 128;
    const int tc = tile / 128;
    const int dt = tc * 64;

    const bf16* arow0 = A + (size_t)(tr * 32 + llo) * D_MODEL;
    const bf16* arow1 = arow0 + 16 * D_MODEL;
    const bf16* wrow0 = Wo + (size_t)(dt + llo) * D_MODEL;

    v8f c[2][4] = {};
    for (int k0 = 0; k0 < D_MODEL; k0 += 32) {
        const int ka = k0 + lhi * 8;
        v16bf a0 = CAT16(*(const v8bf*)(arow0 + ka), *(const v8bf*)(arow0 + ka + 16));
        v16bf a1 = CAT16(*(const v8bf*)(arow1 + ka), *(const v8bf*)(arow1 + ka + 16));
        const int kb = k0 + lhi * 8;
        __builtin_prefetch(wrow0 + k0 + 512, 0, 1);
        for (int n = 0; n < 4; ++n) {
            v16bf b = CAT16(*(const v8bf*)(wrow0 + (size_t)n * 16 * D_MODEL + kb), *(const v8bf*)(wrow0 + (size_t)n * 16 * D_MODEL + kb + 16));
            c[0][n] = __builtin_amdgcn_wmma_f32_16x16x32_bf16(false, a0, false, b,
                                                              (short)0, c[0][n], false, false);
            c[1][n] = __builtin_amdgcn_wmma_f32_16x16x32_bf16(false, a1, false, b,
                                                              (short)0, c[1][n], false, false);
        }
    }
    for (int rr = 0; rr < 2; ++rr)
        for (int n = 0; n < 4; ++n)
            for (int r = 0; r < 8; ++r) {
                const int nrow = tr * 32 + rr * 16 + r + lhi * 8;
                Out[(size_t)nrow * D_MODEL + dt + n * 16 + llo] = c[rr][n][r];
            }
}

// ---------------------------------------------------------------------------
extern "C" void kernel_launch(void* const* d_in, const int* in_sizes, int n_in,
                              void* d_out, int out_size, void* d_ws, size_t ws_size,
                              hipStream_t stream) {
    const float* X    = (const float*)d_in[0];
    const float* Wq   = (const float*)d_in[1];
    const float* Wk   = (const float*)d_in[2];
    const float* Wv   = (const float*)d_in[3];
    const float* Wo   = (const float*)d_in[4];
    const float* cosT = (const float*)d_in[5];
    const float* sinT = (const float*)d_in[6];
    // d_in[7]=position_ids (== arange, folded into RoPE), d_in[8]=mask (causal, folded)

    char* ws = (char*)d_ws;
    const size_t xseg = (size_t)NB * S_LEN * D_MODEL * sizeof(bf16);          // 16.78 MB
    const size_t wseg = (size_t)D_MODEL * D_MODEL * sizeof(bf16);             //  8.39 MB
    bf16* Xb  = (bf16*)(ws);
    bf16* Wqb = (bf16*)(ws + xseg);
    bf16* Wkb = (bf16*)(ws + xseg + wseg);
    bf16* Wvb = (bf16*)(ws + xseg + 2 * wseg);
    bf16* Wob = (bf16*)(ws + xseg + 3 * wseg);
    char* ws2 = ws + xseg + 4 * wseg;
    bf16* Qh = (bf16*)(ws2);
    bf16* Kh = (bf16*)(ws2 + xseg);
    bf16* Vt = (bf16*)(ws2 + 2 * xseg);
    bf16* Ao = (bf16*)(ws2 + 3 * xseg);
    float* Out = (float*)d_out;

    cvt_bf16_kernel<<<4096, 256, 0, stream>>>(X,  Xb);
    cvt_bf16_kernel<<<2048, 256, 0, stream>>>(Wq, Wqb);
    cvt_bf16_kernel<<<2048, 256, 0, stream>>>(Wk, Wkb);
    cvt_bf16_kernel<<<2048, 256, 0, stream>>>(Wv, Wvb);
    cvt_bf16_kernel<<<2048, 256, 0, stream>>>(Wo, Wob);

    qkv_kernel  <<<1536,  256, 0, stream>>>(Xb, Wqb, Wkb, Wvb, Qh, Kh, Vt);
    rope_kernel <<<16384, 256, 0, stream>>>(Qh, Kh, cosT, sinT);
    fa_kernel   <<<1024,  128, 0, stream>>>(Qh, Kh, Vt, Ao);
    oproj_kernel<<<512,   256, 0, stream>>>(Ao, Wob, Out);
}
